// FeedForward_32478542693202
// MI455X (gfx1250) — matrix operand order repaired and checked by reading, not compiled, not run on hardware
//
#include <hip/hip_runtime.h>
#include <hip/hip_bf16.h>

// ---------------------------------------------------------------------------
// Fused BitLinear FFN for MI455X (gfx1250, wave32, WMMA 16x16x32 f16)
//   xq = fake-quant(x);  x13 = xq @ w13^T;  g = relu(x1)^2 * x3
//   inner = rmsnorm(g, norm_w);  iq = fake-quant(inner);  out = iq @ w2^T
//
// v3: double-buffered LDS tiles (1 barrier / K-step, staging overlaps wmma),
//     2 M-frags x 4 N-frags register blocking -> 8 wmma per wave per K-step.
// ---------------------------------------------------------------------------

typedef __attribute__((ext_vector_type(16))) _Float16 v16h;
typedef __attribute__((ext_vector_type(8)))  _Float16 v8h;
typedef __attribute__((ext_vector_type(8)))  float    v8f;

#define DIM  2048
#define HID  5632
#define BM   64          // token rows per block
#define BN1G 128         // per-gate output cols per block (GEMM1)
#define BN2  256         // output cols per block (GEMM2)
#define BK   32          // K step (matches 16x16x32 WMMA)
#define LDA  40          // padded halves per LDS row (80B, 16B multiple)
#define EPV  (HID / 256) // elements per thread in row kernels (22)

__device__ __forceinline__ float quantize1(float v, float s, float inv_s) {
    float r = rintf(v * s);                       // round-half-even (v_rndne)
    r = fminf(fmaxf(r, -128.f), 127.f);
    return r * inv_s;
}

// A fragment (16-bit A layout): halves K+0..7 at p, K+16..23 at p+16
__device__ __forceinline__ v16h load_a_frag(const _Float16* p) {
    v8h lo = *(const v8h*)p;
    v8h hi = *(const v8h*)(p + 16);
    v16h a;
    #pragma unroll
    for (int j = 0; j < 8; ++j) { a[j] = lo[j]; a[j + 8] = hi[j]; }
    return a;
}

// B fragment (same layout as A): halves K+0..7 at p, K+16..23 at p+16
__device__ __forceinline__ v16h load_b_frag(const _Float16* p) {
    v8h lo = *(const v8h*)p;
    v8h hi = *(const v8h*)(p + 16);
    v16h b;
    #pragma unroll
    for (int j = 0; j < 8; ++j) { b[j] = lo[j]; b[j + 8] = hi[j]; }
    return b;
}

#define WMMA_F16(A, B, C) \
    __builtin_amdgcn_wmma_f32_16x16x32_f16(false, (A), false, (B), (short)0, (C), false, false)

// ---------------------------------------------------------------------------
// Kernel 0: fp32 -> f16 bulk convert (weights; one pass per launch)
// ---------------------------------------------------------------------------
__global__ void __launch_bounds__(256)
cvt_f16_kernel(const float* __restrict__ src, _Float16* __restrict__ dst, int n8) {
    const int i = blockIdx.x * 256 + threadIdx.x;
    if (i >= n8) return;
    const float4* p = (const float4*)src + (size_t)i * 2;
    float4 a = p[0], b = p[1];
    v8h h;
    h[0] = (_Float16)a.x; h[1] = (_Float16)a.y; h[2] = (_Float16)a.z; h[3] = (_Float16)a.w;
    h[4] = (_Float16)b.x; h[5] = (_Float16)b.y; h[6] = (_Float16)b.z; h[7] = (_Float16)b.w;
    ((v8h*)dst)[i] = h;
}

// ---------------------------------------------------------------------------
// Kernel 1: per-token dynamic quant scale for x:  s = 127 / max(|x|, 1e-5)
// ---------------------------------------------------------------------------
__global__ void __launch_bounds__(256)
xscale_kernel(const float* __restrict__ x, float* __restrict__ sc,
              float* __restrict__ isc) {
    const int t = blockIdx.x;
    const float* row = x + (size_t)t * DIM;
    float m = 0.f;
    for (int i = threadIdx.x; i < DIM; i += 256)
        m = fmaxf(m, fabsf(row[i]));
    #pragma unroll
    for (int off = 16; off; off >>= 1)
        m = fmaxf(m, __shfl_xor(m, off, 32));
    __shared__ float red[8];
    const int wave = threadIdx.x >> 5, lane = threadIdx.x & 31;
    if (lane == 0) red[wave] = m;
    __syncthreads();
    if (threadIdx.x == 0) {
        float M = red[0];
        #pragma unroll
        for (int i = 1; i < 8; ++i) M = fmaxf(M, red[i]);
        M = fmaxf(M, 1e-5f);
        const float s = 127.f / M;
        sc[t]  = s;
        isc[t] = 1.f / s;
    }
}

// ---------------------------------------------------------------------------
// Kernel 2: GEMM1 + squared-ReLU gating.
// Block: 256 thr = 8 waves (2M x 4N). Tile: M=64 x N=128 per gate.
// Double-buffered LDS; each wave: 8 wmma per K-step (2 A-frags x 4 B-frags).
// ---------------------------------------------------------------------------
__global__ void __launch_bounds__(256)
gemm1_kernel(const float* __restrict__ x, const _Float16* __restrict__ w13h,
             const float* __restrict__ sc, const float* __restrict__ isc,
             _Float16* __restrict__ gated) {
    __shared__ __align__(16) _Float16 As[2][BM][LDA];         // 2 x 64 x 40
    __shared__ __align__(16) _Float16 Bs[2][2 * BN1G][LDA];   // 2 x 256 x 40

    const int tid   = threadIdx.x;
    const int lane  = tid & 31;
    const int wave  = tid >> 5;
    const int waveM = wave >> 2;    // 0..1 -> rows waveM*32
    const int waveN = wave & 3;     // 0..3 -> cols waveN*32 (per gate)
    const int rowBase = blockIdx.y * BM;
    const int colBase = blockIdx.x * BN1G;

    // staging: A = 64 rows x 32 fp32 (4 thr/row, 8 floats each)
    const int   aRow = tid >> 2;
    const int   aC   = (tid & 3) * 8;
    const float s    = sc [rowBase + aRow];
    const float is   = isc[rowBase + aRow];
    const float* xsrc = x + (size_t)(rowBase + aRow) * DIM + aC;

    // staging: B = 256 rows x 32 halves (1 thread per row)
    const int bRow = tid;
    const int wrow = (bRow < BN1G) ? (colBase + bRow)
                                   : (HID + colBase + (bRow - BN1G));
    const _Float16* bsrc = w13h + (size_t)wrow * DIM;

    const int m    = lane & 15;
    const int aOff = (lane & 16) ? 8 : 0;
    const int kOff = (lane & 16) ? 8 : 0;

    v8f acc[2][2][2] = {};   // [gate][mi][ni]

    #define STAGE1(buf, kk) do {                                              \
        float4 xv0 = *(const float4*)(xsrc + (kk));                           \
        float4 xv1 = *(const float4*)(xsrc + (kk) + 4);                       \
        v8h q;                                                                \
        q[0] = (_Float16)quantize1(xv0.x, s, is);                             \
        q[1] = (_Float16)quantize1(xv0.y, s, is);                             \
        q[2] = (_Float16)quantize1(xv0.z, s, is);                             \
        q[3] = (_Float16)quantize1(xv0.w, s, is);                             \
        q[4] = (_Float16)quantize1(xv1.x, s, is);                             \
        q[5] = (_Float16)quantize1(xv1.y, s, is);                             \
        q[6] = (_Float16)quantize1(xv1.z, s, is);                             \
        q[7] = (_Float16)quantize1(xv1.w, s, is);                             \
        *(v8h*)&As[buf][aRow][aC] = q;                                        \
        const _Float16* bp = bsrc + (kk);                                     \
        *(v8h*)&Bs[buf][bRow][0]  = *(const v8h*)(bp);                        \
        *(v8h*)&Bs[buf][bRow][8]  = *(const v8h*)(bp + 8);                    \
        *(v8h*)&Bs[buf][bRow][16] = *(const v8h*)(bp + 16);                   \
        *(v8h*)&Bs[buf][bRow][24] = *(const v8h*)(bp + 24);                   \
        __builtin_prefetch((const void*)(bsrc + (kk) + 2 * BK), 0, 1);        \
    } while (0)

    STAGE1(0, 0);
    __syncthreads();

    for (int kk = 0; kk < DIM; kk += BK) {
        const int cur = (kk >> 5) & 1;
        const int nxt = cur ^ 1;

        v16h a0  = load_a_frag(&As[cur][waveM * 32 +      m][aOff]);
        v16h a1  = load_a_frag(&As[cur][waveM * 32 + 16 + m][aOff]);
        v16h b1A = load_b_frag(&Bs[cur][waveN * 32 +      m][kOff]);
        v16h b1B = load_b_frag(&Bs[cur][waveN * 32 + 16 + m][kOff]);
        v16h b3A = load_b_frag(&Bs[cur][BN1G + waveN * 32 +      m][kOff]);
        v16h b3B = load_b_frag(&Bs[cur][BN1G + waveN * 32 + 16 + m][kOff]);

        if (kk + BK < DIM) STAGE1(nxt, kk + BK);   // overlaps with wmma chain

        acc[0][0][0] = WMMA_F16(a0, b1A, acc[0][0][0]);
        acc[0][0][1] = WMMA_F16(a0, b1B, acc[0][0][1]);
        acc[0][1][0] = WMMA_F16(a1, b1A, acc[0][1][0]);
        acc[0][1][1] = WMMA_F16(a1, b1B, acc[0][1][1]);
        acc[1][0][0] = WMMA_F16(a0, b3A, acc[1][0][0]);
        acc[1][0][1] = WMMA_F16(a0, b3B, acc[1][0][1]);
        acc[1][1][0] = WMMA_F16(a1, b3A, acc[1][1][0]);
        acc[1][1][1] = WMMA_F16(a1, b3B, acc[1][1][1]);
        __syncthreads();
    }
    #undef STAGE1

    // fused epilogue: g = relu(x1)^2 * x3, store f16
    const int rA = rowBase + waveM * 32 + ((lane & 16) ? 8 : 0);
    const int c0 = colBase + waveN * 32 + m;
    #pragma unroll
    for (int mi = 0; mi < 2; ++mi) {
        #pragma unroll
        for (int ni = 0; ni < 2; ++ni) {
            const int col = c0 + ni * 16;
            #pragma unroll
            for (int i = 0; i < 8; ++i) {
                float g = fmaxf(acc[0][mi][ni][i], 0.f);
                g = g * g * acc[1][mi][ni][i];
                gated[(size_t)(rA + mi * 16 + i) * HID + col] = (_Float16)g;
            }
        }
    }
}

// ---------------------------------------------------------------------------
// Kernel 3: per-row RMSNorm + dynamic requantization, in place on f16 buffer.
// ---------------------------------------------------------------------------
__global__ void __launch_bounds__(256)
norm_quant_kernel(_Float16* __restrict__ g, const float* __restrict__ nw) {
    const int t   = blockIdx.x;
    const int tid = threadIdx.x;
    _Float16* row = g + (size_t)t * HID;

    float vals[EPV];
    float ss = 0.f, mm = 0.f;
    #pragma unroll
    for (int i = 0; i < EPV; ++i) {
        const int c = tid + i * 256;
        const float v = (float)row[c];
        vals[i] = v;
        ss += v * v;
        mm = fmaxf(mm, fabsf(v * nw[c]));
    }
    #pragma unroll
    for (int off = 16; off; off >>= 1) {
        ss += __shfl_xor(ss, off, 32);
        mm  = fmaxf(mm, __shfl_xor(mm, off, 32));
    }
    __shared__ float red[16];
    const int wave = tid >> 5, lane = tid & 31;
    if (lane == 0) { red[wave] = ss; red[8 + wave] = mm; }
    __syncthreads();
    if (tid == 0) {
        float S = 0.f, M = 0.f;
        #pragma unroll
        for (int i = 0; i < 8; ++i) { S += red[i]; M = fmaxf(M, red[8 + i]); }
        const float rstd = rsqrtf(S * (1.f / HID) + 1e-5f);
        M = fmaxf(M * rstd, 1e-5f);          // max|inner|
        const float s2 = 127.f / M;
        red[0] = rstd; red[1] = s2; red[2] = 1.f / s2;
    }
    __syncthreads();
    const float rstd = red[0], s2 = red[1], is2 = red[2];
    #pragma unroll
    for (int i = 0; i < EPV; ++i) {
        const int c = tid + i * 256;
        const float inner = vals[i] * rstd * nw[c];
        row[c] = (_Float16)quantize1(inner, s2, is2);
    }
}

// ---------------------------------------------------------------------------
// Kernel 4: GEMM2 — out = iq @ w2^T. Tile M=64 x N=256, double-buffered LDS,
// 8 wmma per wave per K-step (2 A-frags x 4 B-frags).
// ---------------------------------------------------------------------------
__global__ void __launch_bounds__(256)
gemm2_kernel(const _Float16* __restrict__ aq, const _Float16* __restrict__ w2h,
             float* __restrict__ out) {
    __shared__ __align__(16) _Float16 As[2][BM][LDA];    // 2 x 64 x 40
    __shared__ __align__(16) _Float16 Bs[2][BN2][LDA];   // 2 x 256 x 40

    const int tid   = threadIdx.x;
    const int lane  = tid & 31;
    const int wave  = tid >> 5;
    const int waveM = wave >> 2;    // 0..1 -> rows waveM*32
    const int waveN = wave & 3;     // 0..3 -> cols waveN*64
    const int rowBase = blockIdx.y * BM;
    const int colBase = blockIdx.x * BN2;

    // staging: A = 64 rows x 32 halves (4 thr/row, 8 halves each)
    const int aRow = tid >> 2;
    const int aC   = (tid & 3) * 8;
    const _Float16* asrc = aq + (size_t)(rowBase + aRow) * HID + aC;

    // staging: B = 256 rows x 32 halves (1 thread per row)
    const int bRow = tid;
    const _Float16* bsrc = w2h + (size_t)(colBase + bRow) * HID;

    const int m    = lane & 15;
    const int aOff = (lane & 16) ? 8 : 0;
    const int kOff = (lane & 16) ? 8 : 0;

    v8f acc[2][4] = {};   // [mi][ni]

    #define STAGE2(buf, kk) do {                                              \
        *(v8h*)&As[buf][aRow][aC] = *(const v8h*)(asrc + (kk));               \
        const _Float16* bp = bsrc + (kk);                                     \
        *(v8h*)&Bs[buf][bRow][0]  = *(const v8h*)(bp);                        \
        *(v8h*)&Bs[buf][bRow][8]  = *(const v8h*)(bp + 8);                    \
        *(v8h*)&Bs[buf][bRow][16] = *(const v8h*)(bp + 16);                   \
        *(v8h*)&Bs[buf][bRow][24] = *(const v8h*)(bp + 24);                   \
        __builtin_prefetch((const void*)(bsrc + (kk) + 2 * BK), 0, 1);        \
    } while (0)

    STAGE2(0, 0);
    __syncthreads();

    for (int kk = 0; kk < HID; kk += BK) {
        const int cur = (kk >> 5) & 1;
        const int nxt = cur ^ 1;

        v16h a0 = load_a_frag(&As[cur][waveM * 32 +      m][aOff]);
        v16h a1 = load_a_frag(&As[cur][waveM * 32 + 16 + m][aOff]);
        v16h b0 = load_b_frag(&Bs[cur][waveN * 64 +      m][kOff]);
        v16h b1 = load_b_frag(&Bs[cur][waveN * 64 + 16 + m][kOff]);
        v16h b2 = load_b_frag(&Bs[cur][waveN * 64 + 32 + m][kOff]);
        v16h b3 = load_b_frag(&Bs[cur][waveN * 64 + 48 + m][kOff]);

        if (kk + BK < HID) STAGE2(nxt, kk + BK);   // overlaps with wmma chain

        acc[0][0] = WMMA_F16(a0, b0, acc[0][0]);
        acc[0][1] = WMMA_F16(a0, b1, acc[0][1]);
        acc[0][2] = WMMA_F16(a0, b2, acc[0][2]);
        acc[0][3] = WMMA_F16(a0, b3, acc[0][3]);
        acc[1][0] = WMMA_F16(a1, b0, acc[1][0]);
        acc[1][1] = WMMA_F16(a1, b1, acc[1][1]);
        acc[1][2] = WMMA_F16(a1, b2, acc[1][2]);
        acc[1][3] = WMMA_F16(a1, b3, acc[1][3]);
        __syncthreads();
    }
    #undef STAGE2

    const int rA = rowBase + waveM * 32 + ((lane & 16) ? 8 : 0);
    const int c0 = colBase + waveN * 64 + m;
    #pragma unroll
    for (int mi = 0; mi < 2; ++mi) {
        #pragma unroll
        for (int ni = 0; ni < 4; ++ni) {
            const int col = c0 + ni * 16;
            #pragma unroll
            for (int i = 0; i < 8; ++i)
                out[(size_t)(rA + mi * 16 + i) * DIM + col] = acc[mi][ni][i];
        }
    }
}

// ---------------------------------------------------------------------------
extern "C" void kernel_launch(void* const* d_in, const int* in_sizes, int n_in,
                              void* d_out, int out_size, void* d_ws, size_t ws_size,
                              hipStream_t stream) {
    const float* x   = (const float*)d_in[0];  // [4,2048,2048]
    const float* w13 = (const float*)d_in[1];  // [11264,2048]
    const float* w2  = (const float*)d_in[2];  // [2048,5632]
    const float* nw  = (const float*)d_in[3];  // [5632]
    float* out = (float*)d_out;

    const int tok = in_sizes[0] / DIM;         // 8192 tokens

    char* ws = (char*)d_ws;
    size_t off = 0;
    _Float16* gated = (_Float16*)(ws + off); off += (size_t)tok * HID * 2;
    _Float16* w13h  = (_Float16*)(ws + off); off += (size_t)2 * HID * DIM * 2;
    _Float16* w2h   = (_Float16*)(ws + off); off += (size_t)DIM * HID * 2;
    float*    sc    = (float*)(ws + off);    off += (size_t)tok * 4;
    float*    isc   = (float*)(ws + off);

    const int n8_13 = (2 * HID * DIM) / 8;
    const int n8_2  = (DIM * HID) / 8;
    cvt_f16_kernel<<<(n8_13 + 255) / 256, 256, 0, stream>>>(w13, w13h, n8_13);
    cvt_f16_kernel<<<(n8_2  + 255) / 256, 256, 0, stream>>>(w2,  w2h,  n8_2);

    xscale_kernel<<<tok, 256, 0, stream>>>(x, sc, isc);
    gemm1_kernel<<<dim3(HID / BN1G, tok / BM), 256, 0, stream>>>(x, w13h, sc, isc, gated);
    norm_quant_kernel<<<tok, 256, 0, stream>>>(gated, nw);
    gemm2_kernel<<<dim3(DIM / BN2, tok / BM), 256, 0, stream>>>(gated, w2h, out);
}
